// TimeSeriesLSTMEncoder_74062416053162
// MI455X (gfx1250) — matrix operand order repaired and checked by reading, not compiled, not run on hardware
//
#include <hip/hip_runtime.h>
#include <hip/hip_bf16.h>
#include <cstdint>

// ---------------------------------------------------------------------------
// TimeSeriesLSTMEncoder for MI455X (gfx1250, wave32, WMMA, async-LDS copies)
//   Phase 0: pack x / w_ih1 / w_ih2 / w_hh1 / w_hh2 to bf16 (one HBM pass)
//   Phase 1: xg1 = x @ w_ih1^T + b_ih1 + b_hh1   (bf16 WMMA GEMM, async LDS)
//   Phase 2: layer-1 recurrence, persistent single WG, weights in VGPRs;
//            emits out1 directly in bf16
//   Phase 3: xg2 = out1 @ w_ih2^T + b_ih2 + b_hh2 (same GEMM)
//   Phase 4: layer-2 recurrence, keep h_T
//   Phase 5: projection + LayerNorm + tanh head
// ---------------------------------------------------------------------------

typedef __bf16 bf16_t;
typedef __attribute__((ext_vector_type(16))) __bf16 v16bf;
typedef __attribute__((ext_vector_type(8)))  float   v8f;

#define B_   32
#define T_   512
#define DIN  2048
#define H1   256
#define H2   128

// ---------------- CDNA5 async global->LDS copy (ASYNCcnt tracked) ----------

__device__ __forceinline__ void async_b128(uint32_t lds_byte_off, const void* g) {
  // VDST = per-lane LDS byte offset, VADDR = 64-bit global address (GV mode).
  asm volatile("global_load_async_to_lds_b128 %0, %1, off"
               :: "v"(lds_byte_off), "v"((unsigned long long)(uintptr_t)g)
               : "memory");
}
__device__ __forceinline__ void wait_async0() {
  asm volatile("s_wait_asynccnt 0x0" ::: "memory");
}
// Generic pointers to __shared__ carry the 32-bit LDS offset in their low bits.
__device__ __forceinline__ uint32_t lds_off(const void* p) {
  return (uint32_t)(uintptr_t)p;
}

// ---------------- WMMA fragment helpers (layouts per CDNA5 ISA 7.12.2) -----

__device__ __forceinline__ v8f wmma_bf16(v16bf a, v16bf b, v8f c) {
  return __builtin_amdgcn_wmma_f32_16x16x32_bf16(
      /*neg_a=*/false, a, /*neg_b=*/false, b,
      /*c_mod=*/(short)0, c, /*reuse_a=*/false, /*reuse_b=*/false);
}

// A matrix 16x32 bf16, row-major source with leading dim `ld` (halfwords).
__device__ __forceinline__ v16bf load_frag_a(const bf16_t* base, int ld,
                                             int mbase, int kbase) {
  const int lane = threadIdx.x & 31;
  const int half = lane >> 4;
  const bf16_t* p = base + (mbase + (lane & 15)) * ld + kbase + half * 8;
  v16bf r;
#pragma unroll
  for (int i = 0; i < 8; ++i) { r[i] = p[i]; r[i + 8] = p[16 + i]; }
  return r;
}

// B matrix 32x16 bf16 where B[k][n] is stored as W[n][k] (row-major).
__device__ __forceinline__ v16bf load_frag_b(const bf16_t* base, int ld,
                                             int nbase, int kbase) {
  const int lane = threadIdx.x & 31;
  const int half = lane >> 4;
  const bf16_t* p = base + (nbase + (lane & 15)) * ld + kbase + half * 8;
  v16bf r;
#pragma unroll
  for (int i = 0; i < 8; ++i) { r[i] = p[i]; r[i + 8] = p[16 + i]; }
  return r;
}

__device__ __forceinline__ float sigmoidf_(float x) {
  return 1.0f / (1.0f + __expf(-x));
}

// ---------------- f32 -> bf16 pack (4 elements/thread) ---------------------

__global__ void pack_bf16_kernel(const float* __restrict__ in,
                                 bf16_t* __restrict__ out, int n4) {
  const int i = blockIdx.x * blockDim.x + threadIdx.x;
  if (i < n4) {
    const float4 f = ((const float4*)in)[i];
    bf16_t* o = out + i * 4;
    o[0] = (bf16_t)f.x; o[1] = (bf16_t)f.y;
    o[2] = (bf16_t)f.z; o[3] = (bf16_t)f.w;
  }
}

// ---------------- input-projection GEMM (bf16 in, f32 out) -----------------
// C[M,N] = A[M,K]*W[N,K]^T + bias0[N] + bias1[N].
// 256 threads (8 waves), 64x128 tile, K staged 64-wide, double-buffered LDS
// filled with global_load_async_to_lds_b128.

__global__ __launch_bounds__(256)
void gemm_bias_wmma(const bf16_t* __restrict__ A, const bf16_t* __restrict__ W,
                    const float* __restrict__ bias0, const float* __restrict__ bias1,
                    float* __restrict__ C, int K, int N) {
  constexpr int BM = 64, BN = 128, KT = 64, LDT = KT + 8;  // 144B rows: bank-safe
  __shared__ bf16_t As[2][BM * LDT];
  __shared__ bf16_t Ws[2][BN * LDT];

  const int tid  = threadIdx.x;
  const int lane = tid & 31;
  const int wv   = tid >> 5;       // 0..7
  const int m0   = (wv >> 2) * 32; // 0/32
  const int n0   = (wv & 3) * 32;  // 0/32/64/96
  const long rowBase = (long)blockIdx.y * BM;
  const long colBase = (long)blockIdx.x * BN;

  // async stage: A = 64x64 bf16 (512 b128 chunks -> 2/thread),
  //              W = 128x64 bf16 (1024 chunks -> 4/thread)
  auto issue_stage = [&](int buf, int k0) {
#pragma unroll
    for (int q = 0; q < 2; ++q) {
      const int c = tid + 256 * q, r = c >> 3, s = (c & 7) * 8;
      async_b128(lds_off(&As[buf][r * LDT + s]),
                 A + (rowBase + r) * K + k0 + s);
    }
#pragma unroll
    for (int q = 0; q < 4; ++q) {
      const int c = tid + 256 * q, r = c >> 3, s = (c & 7) * 8;
      async_b128(lds_off(&Ws[buf][r * LDT + s]),
                 W + (colBase + r) * K + k0 + s);
    }
  };

  v8f acc[2][2];
#pragma unroll
  for (int a = 0; a < 2; ++a)
#pragma unroll
    for (int b = 0; b < 2; ++b)
#pragma unroll
      for (int i = 0; i < 8; ++i) acc[a][b][i] = 0.0f;

  int buf = 0;
  issue_stage(0, 0);
  wait_async0();
  __syncthreads();

  for (int k0 = 0; k0 < K; k0 += KT) {
    if (k0 + KT < K) issue_stage(buf ^ 1, k0 + KT);
    const bf16_t* as = As[buf];
    const bf16_t* ws = Ws[buf];
#pragma unroll
    for (int kk = 0; kk < 2; ++kk) {
      const v16bf a0 = load_frag_a(as, LDT, m0,      kk * 32);
      const v16bf a1 = load_frag_a(as, LDT, m0 + 16, kk * 32);
      const v16bf b0 = load_frag_b(ws, LDT, n0,      kk * 32);
      const v16bf b1 = load_frag_b(ws, LDT, n0 + 16, kk * 32);
      acc[0][0] = wmma_bf16(a0, b0, acc[0][0]);
      acc[0][1] = wmma_bf16(a0, b1, acc[0][1]);
      acc[1][0] = wmma_bf16(a1, b0, acc[1][0]);
      acc[1][1] = wmma_bf16(a1, b1, acc[1][1]);
    }
    wait_async0();
    __syncthreads();
    buf ^= 1;
  }

  // epilogue: bias + store (C frag: m = r + 8*(lane>=16), n = lane&15)
  const int half = lane >> 4, nl = lane & 15;
#pragma unroll
  for (int mt = 0; mt < 2; ++mt)
#pragma unroll
    for (int nt = 0; nt < 2; ++nt) {
      const long col = colBase + n0 + nt * 16 + nl;
      const float bsum = bias0[col] + bias1[col];
#pragma unroll
      for (int r = 0; r < 8; ++r) {
        const long row = rowBase + m0 + mt * 16 + half * 8 + r;
        C[row * N + col] = acc[mt][nt][r] + bsum;
      }
    }
}

// ---------------- persistent recurrent LSTM kernel -------------------------
// One workgroup of NTH = 4*H threads. Wave w owns gate columns [32w,32w+32).
// Recurrent weights (bf16) live in VGPRs for all 512 steps; h staged in LDS
// as bf16 WMMA fragments; c lives in per-thread registers.

template <int H, int NTH>
__global__ __launch_bounds__(NTH)
void lstm_recurrent(const float* __restrict__ xg,    // [B_,T_,4H] (bias folded)
                    const bf16_t* __restrict__ whh,  // [4H,H] bf16
                    bf16_t* __restrict__ out_seq,    // [B_,T_,H] bf16 or nullptr
                    float* __restrict__ out_hT) {    // [B_,H]    or nullptr
  constexpr int NG  = 4 * H;
  constexpr int LDH = H + 8;    // bf16 halfwords
  constexpr int LDG = NG + 4;   // f32: half-wave rows land in disjoint banks
  constexpr int KS  = H / 32;
  constexpr int TPR = H / 8;
  static_assert(NTH == NG, "one wave per 32 gate columns");

  __shared__ bf16_t h_s[B_ * LDH];
  __shared__ float  g_s[B_ * LDG];

  const int tid   = threadIdx.x;
  const int lane  = tid & 31;
  const int wv    = tid >> 5;
  const int nbase = wv * 32;

  // pin recurrent weight fragments in VGPRs (2 n-tiles x KS k-steps)
  v16bf Bw[2][KS];
#pragma unroll
  for (int nt = 0; nt < 2; ++nt)
#pragma unroll
    for (int k = 0; k < KS; ++k)
      Bw[nt][k] = load_frag_b(whh, H, nbase + nt * 16, k * 32);

  for (int i = tid; i < B_ * LDH; i += NTH) h_s[i] = (bf16_t)0.0f;

  const int eb = tid / TPR;        // batch row for elementwise phase
  const int ej = (tid % TPR) * 8;  // first hidden column
  float c[8];
#pragma unroll
  for (int e = 0; e < 8; ++e) c[e] = 0.0f;

  __syncthreads();

  for (int t = 0; t < T_; ++t) {
    // ---- recurrent GEMM: g = h @ whh^T -------------------------------
    v8f acc[2][2];
#pragma unroll
    for (int a = 0; a < 2; ++a)
#pragma unroll
      for (int b = 0; b < 2; ++b)
#pragma unroll
        for (int i = 0; i < 8; ++i) acc[a][b][i] = 0.0f;
#pragma unroll
    for (int k = 0; k < KS; ++k) {
      const v16bf a0 = load_frag_a(h_s, LDH, 0,  k * 32);
      const v16bf a1 = load_frag_a(h_s, LDH, 16, k * 32);
      acc[0][0] = wmma_bf16(a0, Bw[0][k], acc[0][0]);
      acc[0][1] = wmma_bf16(a0, Bw[1][k], acc[0][1]);
      acc[1][0] = wmma_bf16(a1, Bw[0][k], acc[1][0]);
      acc[1][1] = wmma_bf16(a1, Bw[1][k], acc[1][1]);
    }
    {
      const int half = lane >> 4, nl = lane & 15;
#pragma unroll
      for (int mt = 0; mt < 2; ++mt)
#pragma unroll
        for (int nt = 0; nt < 2; ++nt)
#pragma unroll
          for (int r = 0; r < 8; ++r)
            g_s[(mt * 16 + half * 8 + r) * LDG + nbase + nt * 16 + nl] =
                acc[mt][nt][r];
    }
    __syncthreads();

    // ---- gates + state update ----------------------------------------
    const float* xr = xg + ((long)eb * T_ + t) * NG;
    const float* gr = g_s + eb * LDG;
#pragma unroll
    for (int e = 0; e < 8; ++e) {
      const int j = ej + e;
      const float gi = sigmoidf_(gr[j]         + xr[j]);
      const float gf = sigmoidf_(gr[H + j]     + xr[H + j]);
      const float gg = tanhf    (gr[2 * H + j] + xr[2 * H + j]);
      const float go = sigmoidf_(gr[3 * H + j] + xr[3 * H + j]);
      c[e] = gf * c[e] + gi * gg;
      const float h = go * tanhf(c[e]);
      const bf16_t hb = (bf16_t)h;
      h_s[eb * LDH + j] = hb;
      if (out_seq) out_seq[((long)eb * T_ + t) * H + j] = hb;
      if (out_hT && t == T_ - 1) out_hT[eb * H + j] = h;
    }
    if (t + 1 < T_ && (ej & 63) == 0) {  // hint next step's xg into cache
      const float* nx = xg + ((long)eb * T_ + t + 1) * NG + ej;
      __builtin_prefetch(nx, 0, 1);
      __builtin_prefetch(nx + H, 0, 1);
      __builtin_prefetch(nx + 2 * H, 0, 1);
      __builtin_prefetch(nx + 3 * H, 0, 1);
    }
    __syncthreads();
  }
}

// ---------------- projection + LayerNorm + tanh head ------------------------

__global__ __launch_bounds__(H2)
void head_kernel(const float* __restrict__ hT, const float* __restrict__ Pw,
                 const float* __restrict__ Pb, const float* __restrict__ lnw,
                 const float* __restrict__ lnb, float* __restrict__ out) {
  __shared__ float hrow[H2];
  __shared__ float red[H2];
  const int b = blockIdx.x, j = threadIdx.x;
  hrow[j] = hT[b * H2 + j];
  __syncthreads();
  float acc = Pb[j];
  for (int k = 0; k < H2; ++k) acc += hrow[k] * Pw[j * H2 + k];
  red[j] = acc;
  __syncthreads();
  for (int s = H2 / 2; s > 0; s >>= 1) {
    if (j < s) red[j] += red[j + s];
    __syncthreads();
  }
  const float mu = red[0] * (1.0f / H2);
  __syncthreads();
  const float d = acc - mu;
  red[j] = d * d;
  __syncthreads();
  for (int s = H2 / 2; s > 0; s >>= 1) {
    if (j < s) red[j] += red[j + s];
    __syncthreads();
  }
  const float var = red[0] * (1.0f / H2);
  out[b * H2 + j] = tanhf(d * rsqrtf(var + 1e-5f) * lnw[j] + lnb[j]);
}

// ---------------- launcher --------------------------------------------------

extern "C" void kernel_launch(void* const* d_in, const int* in_sizes, int n_in,
                              void* d_out, int out_size, void* d_ws, size_t ws_size,
                              hipStream_t stream) {
  (void)in_sizes; (void)n_in; (void)out_size; (void)ws_size;
  const float* x      = (const float*)d_in[0];
  const float* w_ih1  = (const float*)d_in[1];
  const float* w_hh1  = (const float*)d_in[2];
  const float* b_ih1  = (const float*)d_in[3];
  const float* b_hh1  = (const float*)d_in[4];
  const float* w_ih2  = (const float*)d_in[5];
  const float* w_hh2  = (const float*)d_in[6];
  const float* b_ih2  = (const float*)d_in[7];
  const float* b_hh2  = (const float*)d_in[8];
  const float* proj_w = (const float*)d_in[9];
  const float* proj_b = (const float*)d_in[10];
  const float* ln_w   = (const float*)d_in[11];
  const float* ln_b   = (const float*)d_in[12];
  float* out = (float*)d_out;

  char* ws = (char*)d_ws;
  size_t off = 0;
  auto alloc = [&](size_t bytes) -> void* {
    void* p = ws + off;
    off = (off + bytes + 255) & ~(size_t)255;
    return p;
  };
  float*  xg1   = (float*)alloc(sizeof(float) * (size_t)B_ * T_ * 4 * H1);
  float*  xg2   = (float*)alloc(sizeof(float) * (size_t)B_ * T_ * 4 * H2);
  float*  hT2   = (float*)alloc(sizeof(float) * (size_t)B_ * H2);
  bf16_t* xb    = (bf16_t*)alloc(sizeof(bf16_t) * (size_t)B_ * T_ * DIN);
  bf16_t* out1b = (bf16_t*)alloc(sizeof(bf16_t) * (size_t)B_ * T_ * H1);
  bf16_t* wih1b = (bf16_t*)alloc(sizeof(bf16_t) * (size_t)4 * H1 * DIN);
  bf16_t* wih2b = (bf16_t*)alloc(sizeof(bf16_t) * (size_t)4 * H2 * H1);
  bf16_t* whh1b = (bf16_t*)alloc(sizeof(bf16_t) * (size_t)4 * H1 * H1);
  bf16_t* whh2b = (bf16_t*)alloc(sizeof(bf16_t) * (size_t)4 * H2 * H2);

  auto pack = [&](const float* src, bf16_t* dst, long n) {
    const int n4 = (int)(n / 4);
    pack_bf16_kernel<<<(n4 + 255) / 256, 256, 0, stream>>>(src, dst, n4);
  };
  pack(x,     xb,    (long)B_ * T_ * DIN);
  pack(w_ih1, wih1b, (long)4 * H1 * DIN);
  pack(w_ih2, wih2b, (long)4 * H2 * H1);
  pack(w_hh1, whh1b, (long)4 * H1 * H1);
  pack(w_hh2, whh2b, (long)4 * H2 * H2);

  {  // xg1 = x @ w_ih1^T + b_ih1 + b_hh1
    dim3 g((4 * H1) / 128, (B_ * T_) / 64);
    gemm_bias_wmma<<<g, 256, 0, stream>>>(xb, wih1b, b_ih1, b_hh1, xg1, DIN, 4 * H1);
  }

  lstm_recurrent<H1, 4 * H1><<<1, 4 * H1, 0, stream>>>(xg1, whh1b, out1b, nullptr);

  {  // xg2 = out1 @ w_ih2^T + b_ih2 + b_hh2
    dim3 g((4 * H2) / 128, (B_ * T_) / 64);
    gemm_bias_wmma<<<g, 256, 0, stream>>>(out1b, wih2b, b_ih2, b_hh2, xg2, H1, 4 * H2);
  }

  lstm_recurrent<H2, 4 * H2><<<1, 4 * H2, 0, stream>>>(xg2, whh2b, nullptr, hT2);

  head_kernel<<<B_, H2, 0, stream>>>(hT2, proj_w, proj_b, ln_w, ln_b, out);
}
